// ChaRNN_61761629716837
// MI455X (gfx1250) — matrix operand order repaired and checked by reading, not compiled, not run on hardware
//
#include <hip/hip_runtime.h>
#include <hip/hip_bf16.h>
#include <math.h>

typedef __attribute__((ext_vector_type(16))) __bf16 bf16x16;
typedef __attribute__((ext_vector_type(8)))  float  f32x8;

namespace {
constexpr int BATCH = 4096;
constexpr int SEQ   = 32;
constexpr int DFEAT = 512;            // per-step image feature width
constexpr int RNN   = 512;
constexpr int NCLS  = 256;
constexpr int GATES = 4 * RNN;        // 2048
constexpr int ALDS  = 40;             // A-tile LDS stride (ushorts): 80B -> 16B-aligned async dsts
constexpr int LDSS  = 36;             // B-tile LDS stride (ushorts): 72B, conflict-free frag reads

union PairPun { unsigned u; __bf16 b[2]; };
union Vec4Pun { uint4 v; unsigned short s[8]; unsigned u[4]; };

__device__ __forceinline__ unsigned short f32_to_bf16_rne(float f) {
  union { float f; unsigned u; } x; x.f = f;
  unsigned r = x.u + 0x7FFFu + ((x.u >> 16) & 1u);
  return (unsigned short)(r >> 16);
}
__device__ __forceinline__ float sigmoidf_(float x) { return 1.0f / (1.0f + __expf(-x)); }

// Per-lane async DMA: 16 bytes global -> LDS, tracked by ASYNCcnt (no VGPR round-trip).
__device__ __forceinline__ void async_copy_b128(unsigned lds_byte_off,
                                                const void* gaddr) {
  asm volatile("global_load_async_to_lds_b128 %0, %1, off"
               :: "v"(lds_byte_off), "v"(gaddr) : "memory");
}
__device__ __forceinline__ void wait_async_zero() {
  asm volatile("s_wait_asynccnt 0" ::: "memory");
}
__device__ __forceinline__ unsigned lds_offset_of(const void* shared_ptr) {
  return (unsigned)(uintptr_t)shared_ptr;   // low 32b of LDS aperture = wave-relative offset
}

__global__ void f32_to_bf16_kernel(const float* __restrict__ src,
                                   unsigned short* __restrict__ dst, int n) {
  for (int i = blockIdx.x * blockDim.x + threadIdx.x; i < n; i += gridDim.x * blockDim.x)
    dst[i] = f32_to_bf16_rne(src[i]);
}
__global__ void zero_u16_kernel(unsigned short* p, int n) {
  for (int i = blockIdx.x * blockDim.x + threadIdx.x; i < n; i += gridDim.x * blockDim.x)
    p[i] = 0;
}
__global__ void zero_f32_kernel(float* p, int n) {
  for (int i = blockIdx.x * blockDim.x + threadIdx.x; i < n; i += gridDim.x * blockDim.x)
    p[i] = 0.0f;
}

// featB[b, :] = bf16(f_pool[b, t, :]) — one step's activation slice, vectorized x4
__global__ void cvt_feat_kernel(const float* __restrict__ f_pool,
                                unsigned short* __restrict__ dst, int t) {
  const int n4 = BATCH * DFEAT / 4;
  for (int i4 = blockIdx.x * blockDim.x + threadIdx.x; i4 < n4;
       i4 += gridDim.x * blockDim.x) {
    int b = i4 >> 7;                 // 128 float4 per 512-wide row
    int k = (i4 & 127) * 4;
    const float4 f = *(const float4*)&f_pool[((size_t)b * SEQ + t) * DFEAT + k];
    uint2 o;
    o.x = (unsigned)f32_to_bf16_rne(f.x) | ((unsigned)f32_to_bf16_rne(f.y) << 16);
    o.y = (unsigned)f32_to_bf16_rne(f.z) | ((unsigned)f32_to_bf16_rne(f.w) << 16);
    *(uint2*)&dst[(size_t)b * DFEAT + k] = o;
  }
}

// One 512-deep K phase of the 4-gate GEMM. A: [BATCH,512] bf16 rows; B: [512,2048] bf16.
__device__ __forceinline__ void lstm_gemm_phase(
    const unsigned short* __restrict__ Asrc,
    const unsigned short* __restrict__ Bsrc,
    int m0, int n0, int tid, int lane, int wm, int wn,
    unsigned short* __restrict__ As,           // [32*ALDS]
    unsigned short* __restrict__ Bs,           // [4][32*LDSS] flat, [gate][col][k]
    f32x8* __restrict__ acc)
{
  const int half = lane >> 4;
  const int arow = 16 * wm + (lane & 15);
  const int bcol = 16 * wn + (lane & 15);
  const int arw  = tid >> 2;                   // A stage row
  const int ako  = (tid & 3) * 8;              // A stage k offset
  const unsigned a_lds = lds_offset_of(&As[arw * ALDS + ako]);   // 16B-aligned

  for (int kk = 0; kk < 512; kk += 32) {
    // stage A: per-lane async DMA of 16B (8 bf16) straight into LDS
    async_copy_b128(a_lds, &Asrc[(size_t)(m0 + arw) * DFEAT + kk + ako]);

    // stage B: 4 coalesced uint4 loads/thread, scatter-transpose into [gate][col][k]
#pragma unroll
    for (int i = 0; i < 4; ++i) {
      int g    = tid + 128 * i;
      int gate = g >> 7;
      int k    = (g >> 2) & 31;
      int c8   = (g & 3) * 8;
      const unsigned short* src =
          &Bsrc[(size_t)(kk + k) * GATES + n0 + (gate << 9) + c8];
      Vec4Pun bv; bv.v = *(const uint4*)src;
      if (kk + 32 < 512)
        __builtin_prefetch(src + (size_t)32 * GATES, 0, 1);
#pragma unroll
      for (int j = 0; j < 8; ++j)
        Bs[gate * (32 * LDSS) + (c8 + j) * LDSS + k] = bv.s[j];
    }
    wait_async_zero();          // A-tile DMA complete (ASYNCcnt)
    __syncthreads();            // DS stores + cross-wave visibility

    // A fragment: CDNA5 16-bit A 16x32 layout (lane=M, pairwise K with 8-halves)
    bf16x16 af;
#pragma unroll
    for (int v = 0; v < 8; ++v) {
      int k = 2 * v + 8 * half + (v >= 4 ? 8 : 0);
      PairPun p; p.u = *(const unsigned*)&As[arow * ALDS + k];
      af[2 * v] = p.b[0]; af[2 * v + 1] = p.b[1];
    }
#pragma unroll
    for (int gate = 0; gate < 4; ++gate) {
      // B fragment: 16-bit B 32x16 layout (lane=N, pairwise K with 8-halves, matching A)
      bf16x16 bf;
#pragma unroll
      for (int j = 0; j < 8; ++j) {
        int k = 2 * j + 8 * half + (j >= 4 ? 8 : 0);
        PairPun p; p.u = *(const unsigned*)&Bs[gate * (32 * LDSS) + bcol * LDSS + k];
        bf[2 * j] = p.b[0]; bf[2 * j + 1] = p.b[1];
      }
      acc[gate] = __builtin_amdgcn_wmma_f32_16x16x32_bf16(false, af, false, bf,
                                                          (short)0, acc[gate],
                                                          false, false);
    }
    __syncthreads();
  }
}

// One LSTM step: z = [feat|h] @ [Wx(:512)|Wh] + bias + Wx[512+gt_prev] gather; gates; c/h.
__global__ __launch_bounds__(128)
void charnn_lstm_step(const unsigned short* __restrict__ featB, // [B,512] bf16
                      const unsigned short* __restrict__ h_in,  // [B,512] bf16
                      const unsigned short* __restrict__ WxB,   // [512,2048] bf16
                      const unsigned short* __restrict__ WhB,   // [512,2048] bf16
                      const int*   __restrict__ gtruth,         // [B,32]
                      const float* __restrict__ Wx_full,        // [768,2048] f32
                      const float* __restrict__ bias,           // [2048] f32
                      unsigned short* __restrict__ h_outB,      // [B,512] bf16
                      float* __restrict__ h_outF,               // [B,512] f32
                      float* __restrict__ c_state,              // [B,512] f32 in/out
                      int t)
{
  __shared__ unsigned short As[32 * ALDS];
  __shared__ unsigned short Bs[4 * 32 * LDSS];

  const int tid  = threadIdx.x;
  const int lane = tid & 31;
  const int wave = tid >> 5;
  const int wm = wave >> 1, wn = wave & 1;
  const int m0 = blockIdx.y * 32;
  const int n0 = blockIdx.x * 32;

  f32x8 acc[4];
#pragma unroll
  for (int g = 0; g < 4; ++g)
#pragma unroll
    for (int i = 0; i < 8; ++i) acc[g][i] = 0.0f;

  lstm_gemm_phase(featB, WxB, m0, n0, tid, lane, wm, wn, As, Bs, acc);  // x @ Wx
  lstm_gemm_phase(h_in,  WhB, m0, n0, tid, lane, wm, wn, As, Bs, acc);  // h @ Wh

  const int half  = lane >> 4;
  const int colin = lane & 15;
#pragma unroll
  for (int v = 0; v < 8; ++v) {
    int gr = m0 + 16 * wm + v + 8 * half;
    int rc = n0 + 16 * wn + colin;
    float addi = bias[rc], addf = bias[rc + RNN];
    float addg = bias[rc + 2 * RNN], addo = bias[rc + 3 * RNN];
    if (t > 0) {
      int prev = gtruth[gr * SEQ + (t - 1)];
      const float* wrow = Wx_full + (size_t)(DFEAT + prev) * GATES;
      addi += wrow[rc];           addf += wrow[rc + RNN];
      addg += wrow[rc + 2 * RNN]; addo += wrow[rc + 3 * RNN];
    }
    float zi = acc[0][v] + addi;
    float zf = acc[1][v] + addf;
    float zg = acc[2][v] + addg;
    float zo = acc[3][v] + addo;
    float ig = sigmoidf_(zi), fg = sigmoidf_(zf);
    float gg = tanhf(zg),     og = sigmoidf_(zo);
    size_t idx = (size_t)gr * RNN + rc;
    float cn = fg * c_state[idx] + ig * gg;
    float hn = og * tanhf(cn);
    c_state[idx] = cn;
    h_outF[idx]  = hn;
    h_outB[idx]  = f32_to_bf16_rne(hn);
  }
}

// logits[:, t, :] = h @ softmax_w + softmax_b   (M=4096, N=256, K=512)
__global__ __launch_bounds__(128)
void charnn_logits(const unsigned short* __restrict__ hB,   // [B,512] bf16
                   const unsigned short* __restrict__ swB,  // [512,256] bf16
                   const float* __restrict__ sbias,         // [256] f32
                   float* __restrict__ seq_out,             // [B,T,C] f32
                   int t)
{
  __shared__ unsigned short As[32 * ALDS];
  __shared__ unsigned short Bs[32 * LDSS];   // [col][k]

  const int tid  = threadIdx.x;
  const int lane = tid & 31;
  const int wave = tid >> 5;
  const int wm = wave >> 1, wn = wave & 1;
  const int m0 = blockIdx.y * 32;
  const int n0 = blockIdx.x * 32;

  f32x8 acc;
#pragma unroll
  for (int i = 0; i < 8; ++i) acc[i] = 0.0f;

  const int half = lane >> 4;
  const int arow = 16 * wm + (lane & 15);
  const int bcol = 16 * wn + (lane & 15);
  const int arw  = tid >> 2;
  const int ako  = (tid & 3) * 8;
  const unsigned a_lds = lds_offset_of(&As[arw * ALDS + ako]);

  for (int kk = 0; kk < RNN; kk += 32) {
    async_copy_b128(a_lds, &hB[(size_t)(m0 + arw) * RNN + kk + ako]);
    {
      int k  = tid >> 2;
      int c8 = (tid & 3) * 8;
      Vec4Pun bv; bv.v = *(const uint4*)&swB[(size_t)(kk + k) * NCLS + n0 + c8];
#pragma unroll
      for (int j = 0; j < 8; ++j)
        Bs[(c8 + j) * LDSS + k] = bv.s[j];
    }
    wait_async_zero();
    __syncthreads();

    bf16x16 af;
#pragma unroll
    for (int v = 0; v < 8; ++v) {
      int k = 2 * v + 8 * half + (v >= 4 ? 8 : 0);
      PairPun p; p.u = *(const unsigned*)&As[arow * ALDS + k];
      af[2 * v] = p.b[0]; af[2 * v + 1] = p.b[1];
    }
    bf16x16 bf;
#pragma unroll
    for (int j = 0; j < 8; ++j) {
      int k = 2 * j + 8 * half + (j >= 4 ? 8 : 0);
      PairPun p; p.u = *(const unsigned*)&Bs[bcol * LDSS + k];
      bf[2 * j] = p.b[0]; bf[2 * j + 1] = p.b[1];
    }
    acc = __builtin_amdgcn_wmma_f32_16x16x32_bf16(false, af, false, bf,
                                                  (short)0, acc, false, false);
    __syncthreads();
  }

  const int colin = lane & 15;
#pragma unroll
  for (int v = 0; v < 8; ++v) {
    int gr = m0 + 16 * wm + v + 8 * half;
    int gc = n0 + 16 * wn + colin;
    seq_out[(size_t)gr * (SEQ * NCLS) + (size_t)t * NCLS + gc] = acc[v] + sbias[gc];
  }
}
} // anonymous namespace

extern "C" void kernel_launch(void* const* d_in, const int* in_sizes, int n_in,
                              void* d_out, int out_size, void* d_ws, size_t ws_size,
                              hipStream_t stream) {
  (void)in_sizes; (void)n_in; (void)out_size; (void)ws_size;

  const float* f_pool = (const float*)d_in[0];
  const int*   gt     = (const int*)d_in[1];
  const float* Wx     = (const float*)d_in[2]; // [768, 2048]
  const float* Wh     = (const float*)d_in[3]; // [512, 2048]
  const float* bias   = (const float*)d_in[4]; // [2048]
  const float* sw     = (const float*)d_in[5]; // [512, 256]
  const float* sb     = (const float*)d_in[6]; // [256]

  float* seq  = (float*)d_out;                                  // [B,T,C]
  float* hOut = seq  + (size_t)BATCH * SEQ * NCLS;              // [B,R]
  float* cOut = hOut + (size_t)BATCH * RNN;                     // [B,R] state

  char* ws = (char*)d_ws;
  unsigned short* WxB   = (unsigned short*)ws; ws += (size_t)DFEAT * GATES * 2;
  unsigned short* WhB   = (unsigned short*)ws; ws += (size_t)RNN   * GATES * 2;
  unsigned short* swB   = (unsigned short*)ws; ws += (size_t)RNN   * NCLS  * 2;
  unsigned short* hb0   = (unsigned short*)ws; ws += (size_t)BATCH * RNN   * 2;
  unsigned short* hb1   = (unsigned short*)ws; ws += (size_t)BATCH * RNN   * 2;
  unsigned short* featB = (unsigned short*)ws;                  // total ~16.25 MiB

  f32_to_bf16_kernel<<<512, 256, 0, stream>>>(Wx, WxB, DFEAT * GATES);
  f32_to_bf16_kernel<<<512, 256, 0, stream>>>(Wh, WhB, RNN * GATES);
  f32_to_bf16_kernel<<<128, 256, 0, stream>>>(sw, swB, RNN * NCLS);
  zero_u16_kernel<<<512, 256, 0, stream>>>(hb0, BATCH * RNN);   // h0 = 0
  zero_f32_kernel<<<512, 256, 0, stream>>>(cOut, BATCH * RNN);  // c0 = 0

  dim3 gs(RNN / 32, BATCH / 32);   // (16, 128)
  dim3 gl(NCLS / 32, BATCH / 32);  // (8, 128)
  for (int t = 0; t < SEQ; ++t) {
    const unsigned short* hin = (t & 1) ? hb1 : hb0;   // ping-pong (race-free)
    unsigned short*      hout = (t & 1) ? hb0 : hb1;
    cvt_feat_kernel<<<1024, 256, 0, stream>>>(f_pool, featB, t);
    charnn_lstm_step<<<gs, 128, 0, stream>>>(featB, hin, WxB, WhB, gt, Wx, bias,
                                             hout, hOut, cOut, t);
    charnn_logits<<<gl, 128, 0, stream>>>(hout, swB, sb, seq, t);
  }
}
